// IncrementalClassRectificationLoss_33732673143133
// MI455X (gfx1250) — compile-verified, matrix operand order checked by reading, not run on hardware
//
#include <hip/hip_runtime.h>
#include <hip/hip_bf16.h>
#include <float.h>
#include <limits.h>
#include <math.h>

#define B_ROWS   4096
#define C_CLS    2000
#define KTOP     3
#define MARGIN_F 1.0f
#define ALPHA_F  0.001f
#define CH       8                       // row chunks for stats pass
#define ROWS_PER_CH (B_ROWS / CH)        // 512
#define ABLK_Y   16                      // row chunks for anchor pass
#define ROWS_PER_ABLK (B_ROWS / ABLK_Y)  // 256

typedef __attribute__((ext_vector_type(16))) _Float16 v16h;
typedef __attribute__((ext_vector_type(8)))  float    v8f;

// ---------------------------------------------------------------------------
// Kernel 1: per-class positive counts via WMMA.  counts[c] = sum_b target[b,c]
// A = 16x32 all-ones (f16)  ->  D[m][n] = sum_k B[k][n], layout-robust
// (all-ones A makes the result invariant to the K-permutation of the B frag).
// One wave per 16-class tile, 64 rows per iteration with TWO independent
// accumulators so consecutive v_wmma ops have no C-operand RAW chain
// (CDNA5 WMMA->WMMA RAW costs ~5 slots; dual accumulators pipeline instead).
// target in {0,1} is exact in f16; accumulation is f32.
// ---------------------------------------------------------------------------
__global__ void icrl_counts_wmma(const float* __restrict__ tgt,
                                 float* __restrict__ counts) {
    const int lane   = threadIdx.x;           // blockDim.x == 32 (one wave)
    const int c0     = blockIdx.x * 16;
    const int col    = c0 + (lane & 15);
    const int rowAdd = (lane >= 16) ? 8 : 0;  // lane-half K split (16-bit B frag)

    v16h a;
#pragma unroll
    for (int e = 0; e < 16; ++e) a[e] = (_Float16)1.0f;

    v8f acc0 = {};
    v8f acc1 = {};
    for (int row0 = 0; row0 < B_ROWS; row0 += 64) {
        v16h b0, b1;
#pragma unroll
        for (int e = 0; e < 16; ++e) {
            const int k = (e < 8) ? (e + rowAdd) : (e + 8 + rowAdd);
            b0[e] = (_Float16)tgt[(size_t)(row0 + k) * C_CLS + col];
            b1[e] = (_Float16)tgt[(size_t)(row0 + 32 + k) * C_CLS + col];
        }
        acc0 = __builtin_amdgcn_wmma_f32_16x16x32_f16(
                   false, a, false, b0, (short)0, acc0, false, false);
        acc1 = __builtin_amdgcn_wmma_f32_16x16x32_f16(
                   false, a, false, b1, (short)0, acc1, false, false);
    }
    if (lane < 16) counts[c0 + lane] = acc0[0] + acc1[0];  // M=0 row, N=lane
}

// ---------------------------------------------------------------------------
// Kernel 2: streaming stats. Thread = one class, block.y = row chunk.
// Produces per-chunk: top-4 smallest positive (val,row), top-3 smallest
// negative vals, and BCE partial sum (double -> one f32 atomic per block).
// Coalesced: 32 lanes read 32 consecutive f32 per row step.
// ---------------------------------------------------------------------------
__global__ void icrl_stats_partial(const float* __restrict__ inp,
                                   const float* __restrict__ tgt,
                                   float* __restrict__ accum,
                                   float* __restrict__ ppv,
                                   int*   __restrict__ ppi,
                                   float* __restrict__ pnv) {
    const int tid   = threadIdx.x;
    const int c     = blockIdx.x * blockDim.x + tid;
    const int chunk = blockIdx.y;
    const int b0    = chunk * ROWS_PER_CH;

    double bce = 0.0;
    float pv[4] = {INFINITY, INFINITY, INFINITY, INFINITY};
    int   pi[4] = {INT_MAX, INT_MAX, INT_MAX, INT_MAX};
    float nv[3] = {INFINITY, INFINITY, INFINITY};

    if (c < C_CLS) {
        for (int b = b0; b < b0 + ROWS_PER_CH; ++b) {
            const size_t idx = (size_t)b * C_CLS + c;
            if (b + 16 < b0 + ROWS_PER_CH) {               // stream-ahead hints
                __builtin_prefetch(&inp[idx + 16 * (size_t)C_CLS], 0, 1);
                __builtin_prefetch(&tgt[idx + 16 * (size_t)C_CLS], 0, 1);
            }
            const float x = inp[idx];
            const float t = tgt[idx];
            // logaddexp(0,x) - x*t  =  max(x,0) + log1p(exp(-|x|)) - x*t
            bce += (double)(fmaxf(x, 0.0f) + log1pf(expf(-fabsf(x))) - x * t);
            const float p = 1.0f / (1.0f + expf(-x));
            if (t == 1.0f) {
                if (p < pv[3]) {             // strict < keeps stable tie order
                    int q = 3;
                    while (q > 0 && p < pv[q - 1]) {
                        pv[q] = pv[q - 1]; pi[q] = pi[q - 1]; --q;
                    }
                    pv[q] = p; pi[q] = b;
                }
            } else {
                if (p < nv[2]) {
                    int q = 2;
                    while (q > 0 && p < nv[q - 1]) { nv[q] = nv[q - 1]; --q; }
                    nv[q] = p;
                }
            }
        }
        const size_t b4 = ((size_t)chunk * C_CLS + c) * 4;
        const size_t b3 = ((size_t)chunk * C_CLS + c) * 3;
#pragma unroll
        for (int j = 0; j < 4; ++j) { ppv[b4 + j] = pv[j]; ppi[b4 + j] = pi[j]; }
#pragma unroll
        for (int l = 0; l < 3; ++l) pnv[b3 + l] = nv[l];
    }

    __shared__ double red[256];
    red[tid] = bce;
    __syncthreads();
    for (int s = 128; s > 0; s >>= 1) {
        if (tid < s) red[tid] += red[tid + s];
        __syncthreads();
    }
    if (tid == 0) atomicAdd(&accum[0], (float)red[0]);
}

// ---------------------------------------------------------------------------
// Kernel 3: merge per-chunk partial top-k lists -> final per-class lists.
// Lexicographic (value,row) merge == global stable sort order.
// ---------------------------------------------------------------------------
__global__ void icrl_stats_merge(const float* __restrict__ ppv,
                                 const int*   __restrict__ ppi,
                                 const float* __restrict__ pnv,
                                 float* __restrict__ fpv,
                                 int*   __restrict__ fpi,
                                 float* __restrict__ fnv) {
    const int c = blockIdx.x * blockDim.x + threadIdx.x;
    if (c >= C_CLS) return;

    float mv[4] = {INFINITY, INFINITY, INFINITY, INFINITY};
    int   mi[4] = {INT_MAX, INT_MAX, INT_MAX, INT_MAX};
    float nv[3] = {INFINITY, INFINITY, INFINITY};

    for (int ch = 0; ch < CH; ++ch) {
        const size_t b4 = ((size_t)ch * C_CLS + c) * 4;
        const size_t b3 = ((size_t)ch * C_CLS + c) * 3;
#pragma unroll
        for (int j = 0; j < 4; ++j) {
            const float v  = ppv[b4 + j];
            const int   id = ppi[b4 + j];
            if (v < mv[3] || (v == mv[3] && id < mi[3])) {
                int q = 3;
                while (q > 0 && (v < mv[q - 1] || (v == mv[q - 1] && id < mi[q - 1]))) {
                    mv[q] = mv[q - 1]; mi[q] = mi[q - 1]; --q;
                }
                mv[q] = v; mi[q] = id;
            }
        }
#pragma unroll
        for (int l = 0; l < 3; ++l) {
            const float v = pnv[b3 + l];
            if (v < nv[2]) {
                int q = 2;
                while (q > 0 && v < nv[q - 1]) { nv[q] = nv[q - 1]; --q; }
                nv[q] = v;
            }
        }
    }
#pragma unroll
    for (int j = 0; j < 4; ++j) {
        fpv[(size_t)c * 4 + j] = (mv[j] < 3.0e38f) ? mv[j] : 0.0f;  // INF -> 0
        fpi[(size_t)c * 4 + j] = mi[j];
    }
#pragma unroll
    for (int l = 0; l < 3; ++l)
        fnv[(size_t)c * 3 + l] = (nv[l] < 3.0e38f) ? nv[l] : 0.0f;
}

// ---------------------------------------------------------------------------
// Kernel 4: minority mask.  Bitonic sort of 2048 (count,class) pairs in LDS
// (lexicographic => equals stable argsort), inclusive scan, select.
// ---------------------------------------------------------------------------
__global__ void icrl_minority(const float* __restrict__ counts,
                              int* __restrict__ minority) {
    const int N = 2048;
    __shared__ float sv[N];
    __shared__ int   si[N];
    __shared__ float cs[N];
    const int tid = threadIdx.x;            // blockDim.x == 1024

    for (int i = tid; i < N; i += 1024) {
        if (i < C_CLS) { sv[i] = counts[i]; si[i] = i; }
        else           { sv[i] = INFINITY;  si[i] = i; }
    }
    __syncthreads();

    for (int k = 2; k <= N; k <<= 1) {
        for (int j = k >> 1; j > 0; j >>= 1) {
            for (int i = tid; i < N; i += 1024) {
                const int ixj = i ^ j;
                if (ixj > i) {
                    const bool up = ((i & k) == 0);
                    const float a = sv[i], b = sv[ixj];
                    const int  ai = si[i], bi = si[ixj];
                    const bool agtb = (a > b) || (a == b && ai > bi);
                    if (up == agtb) {
                        sv[i] = b;  si[i] = bi;
                        sv[ixj] = a; si[ixj] = ai;
                    }
                }
            }
            __syncthreads();
        }
    }

    for (int i = tid; i < N; i += 1024) cs[i] = (sv[i] < 3.0e38f) ? sv[i] : 0.0f;
    __syncthreads();
    for (int off = 1; off < N; off <<= 1) {
        const int i1 = tid + 1024;
        const float a0 = (tid >= off) ? cs[tid - off] : 0.0f;
        const float a1 = (i1  >= off) ? cs[i1 - off]  : 0.0f;
        __syncthreads();
        cs[tid] += a0;
        cs[i1]  += a1;
        __syncthreads();
    }
    for (int i = tid; i < N; i += 1024) {
        if (si[i] < C_CLS) {
            minority[si[i]] = (cs[i] <= 0.5f * (float)B_ROWS && sv[i] > 1.0f) ? 1 : 0;
        }
    }
}

// ---------------------------------------------------------------------------
// Kernel 5: anchor pass.  Thread = class, block.y = row chunk; only minority
// classes do work (L2-resident re-read).  Rank of an anchor is its slot in
// the stored top-4 (row-index match) or >K otherwise.
// ---------------------------------------------------------------------------
__global__ void icrl_anchor(const float* __restrict__ inp,
                            const float* __restrict__ tgt,
                            const float* __restrict__ counts,
                            const int*   __restrict__ minority,
                            const float* __restrict__ fpv,
                            const int*   __restrict__ fpi,
                            const float* __restrict__ fnv,
                            float* __restrict__ accum) {
    const int tid = threadIdx.x;
    const int c   = blockIdx.x * blockDim.x + tid;
    const int b0  = blockIdx.y * ROWS_PER_ABLK;

    float dpl = 0.0f, dnl = 0.0f;
    if (c < C_CLS && minority[c]) {
        const int n_pos = (int)(counts[c] + 0.5f);
        const int m   = min(KTOP, max(n_pos - 1, 0));
        const int n_n = min(KTOP, B_ROWS - n_pos);
        float pval[4]; int pidx[4]; float nval[3];
#pragma unroll
        for (int j = 0; j < 4; ++j) {
            pval[j] = fpv[(size_t)c * 4 + j];
            pidx[j] = fpi[(size_t)c * 4 + j];
        }
#pragma unroll
        for (int l = 0; l < 3; ++l) nval[l] = fnv[(size_t)c * 3 + l];

        for (int b = b0; b < b0 + ROWS_PER_ABLK; ++b) {
            const size_t idx = (size_t)b * C_CLS + c;
            if (tgt[idx] == 1.0f) {
                const float p = 1.0f / (1.0f + expf(-inp[idx]));
                int r = KTOP + 1;
#pragma unroll
                for (int j = 0; j < 4; ++j) if (pidx[j] == b) r = j;
                float dpos = 0.0f;
                if (r > KTOP) {
                    for (int j = 0; j < m; ++j) dpos += fabsf(p - pval[j]);
                } else {
                    int pa = 0;
#pragma unroll
                    for (int j = 0; j <= KTOP; ++j) {
                        if (j != r) {
                            if (pa < m) dpos += fabsf(p - pval[j]);
                            ++pa;
                        }
                    }
                }
                float dneg = 0.0f;
                for (int l = 0; l < n_n; ++l) dneg += fabsf(p - nval[l]);
                dpl += (float)n_n * dpos;
                dnl += (float)m   * dneg;
            }
        }
    }

    __shared__ float r1[256];
    __shared__ float r2[256];
    r1[tid] = dpl; r2[tid] = dnl;
    __syncthreads();
    for (int s = 128; s > 0; s >>= 1) {
        if (tid < s) { r1[tid] += r1[tid + s]; r2[tid] += r2[tid + s]; }
        __syncthreads();
    }
    if (tid == 0) {
        if (r1[0] != 0.0f) atomicAdd(&accum[1], r1[0]);
        if (r2[0] != 0.0f) atomicAdd(&accum[2], r2[0]);
    }
}

// ---------------------------------------------------------------------------
// Kernel 6: finalize scalar loss.
// ---------------------------------------------------------------------------
__global__ void icrl_finalize(const float* __restrict__ accum,
                              float* __restrict__ out) {
    const float bce = accum[0] / (float)((size_t)B_ROWS * (size_t)C_CLS);
    const float crl = fmaxf(accum[1] - accum[2] + MARGIN_F, 0.0f);
    out[0] = ALPHA_F * crl + (1.0f - ALPHA_F) * bce;
}

// ---------------------------------------------------------------------------
// Workspace layout (bytes, all 16B aligned):
//   [0,16)           accum: bce, dp, dn, pad
//   [16, 8016)       counts            C f32
//   [8016, 16016)    minority          C i32
//   [16016, 48016)   fpv               C*4 f32
//   [48016, 80016)   fpi               C*4 i32
//   [80016, 104016)  fnv               C*3 f32
//   [104016,360016)  ppv               CH*C*4 f32
//   [360016,616016)  ppi               CH*C*4 i32
//   [616016,808016)  pnv               CH*C*3 f32
// ---------------------------------------------------------------------------
extern "C" void kernel_launch(void* const* d_in, const int* in_sizes, int n_in,
                              void* d_out, int out_size, void* d_ws, size_t ws_size,
                              hipStream_t stream) {
    const float* inp = (const float*)d_in[0];
    const float* tgt = (const float*)d_in[1];
    // d_in[2] (X) is unused by the reference loss.
    float* out = (float*)d_out;

    char* ws = (char*)d_ws;
    float* accum    = (float*)(ws + 0);
    float* counts   = (float*)(ws + 16);
    int*   minority = (int*)  (ws + 8016);
    float* fpv      = (float*)(ws + 16016);
    int*   fpi      = (int*)  (ws + 48016);
    float* fnv      = (float*)(ws + 80016);
    float* ppv      = (float*)(ws + 104016);
    int*   ppi      = (int*)  (ws + 360016);
    float* pnv      = (float*)(ws + 616016);

    hipMemsetAsync(accum, 0, 16, stream);

    icrl_counts_wmma<<<C_CLS / 16, 32, 0, stream>>>(tgt, counts);
    icrl_stats_partial<<<dim3((C_CLS + 255) / 256, CH), 256, 0, stream>>>(
        inp, tgt, accum, ppv, ppi, pnv);
    icrl_stats_merge<<<(C_CLS + 255) / 256, 256, 0, stream>>>(
        ppv, ppi, pnv, fpv, fpi, fnv);
    icrl_minority<<<1, 1024, 0, stream>>>(counts, minority);
    icrl_anchor<<<dim3((C_CLS + 255) / 256, ABLK_Y), 256, 0, stream>>>(
        inp, tgt, counts, minority, fpv, fpi, fnv, accum);
    icrl_finalize<<<1, 1, 0, stream>>>(accum, out);
}
